// MulLayer_90546500534708
// MI455X (gfx1250) — matrix operand order repaired and checked by reading, not compiled, not run on hardware
//
#include <hip/hip_runtime.h>

// ---------------------------------------------------------------------------
// MI455X (gfx1250, wave32) implementation of the MulLayer style-transfer op.
// bf16 WMMA (v_wmma_f32_16x16x32_bf16) for: 3x3 convs (implicit GEMM),
// masked covariances, compress GEMM, and the fused transform+unzip epilogue.
// Activations are channel-last so global->LDS staging is contiguous on both
// sides and uses gfx1250 async-to-LDS copies (ASYNCcnt).
// ---------------------------------------------------------------------------

typedef __bf16 bf16_t;
typedef __attribute__((ext_vector_type(16))) __bf16 v16bf;
typedef __attribute__((ext_vector_type(8)))  __bf16 v8bf;
typedef __attribute__((ext_vector_type(8)))  float  v8f;
typedef int v4i_ __attribute__((vector_size(16)));

#define HH   256
#define WW   256
#define HWP  65536
#define CC   512
#define MDIM 32
#define NMAXM 9

#if defined(__HIP_DEVICE_COMPILE__) && \
    __has_builtin(__builtin_amdgcn_global_load_async_to_lds_b128) && \
    __has_builtin(__builtin_amdgcn_s_wait_asynccnt)
#define USE_ASYNC_LDS 1
#else
#define USE_ASYNC_LDS 0
#endif

static __device__ __forceinline__ void copy16_g2l(const bf16_t* src, bf16_t* dst) {
#if USE_ASYNC_LDS
  __builtin_amdgcn_global_load_async_to_lds_b128(
      (__attribute__((address_space(1))) v4i_*)src,
      (__attribute__((address_space(3))) v4i_*)dst, 0, 0);
#else
  *(v8bf*)dst = *(const v8bf*)src;
#endif
}

static __device__ __forceinline__ void async_wait_all() {
#if USE_ASYNC_LDS
  __builtin_amdgcn_s_wait_asynccnt(0);
#endif
}

static __device__ __forceinline__ v8f wmma_bf16(v16bf a, v16bf b, v8f c) {
  return __builtin_amdgcn_wmma_f32_16x16x32_bf16(false, a, false, b, (short)0, c,
                                                 false, false);
}

// ---------------------------------------------------------------------------
// Utility kernels
// ---------------------------------------------------------------------------
__global__ void k_zerof(float* p, int n) {
  int i = blockIdx.x * 256 + threadIdx.x;
  if (i < n) p[i] = 0.f;
}

__global__ void k_cast(const float* __restrict__ a, bf16_t* __restrict__ b, int n) {
  int i = blockIdx.x * 256 + threadIdx.x;
  if (i < n) b[i] = (bf16_t)a[i];
}

// conv weight repack: [OC][IC][3][3] f32 -> [OC][3][3][IC] bf16
__global__ void k_packconv(const float* __restrict__ w, bf16_t* __restrict__ wt,
                           int OC, int IC) {
  long idx = (long)blockIdx.x * 256 + threadIdx.x;
  long total = (long)OC * IC * 9;
  if (idx >= total) return;
  int ic = (int)(idx % IC);
  long r  = idx / IC;
  int kk = (int)(r % 9);
  int oc = (int)(r / 9);
  wt[idx] = (bf16_t)w[((long)oc * IC + ic) * 9 + kk];
}

// pack mask membership bits per pixel: bit i set if mask_i[p] > 0
__global__ void k_maskbits(const int* __restrict__ cm, const int* __restrict__ sm,
                           unsigned* __restrict__ mbC, unsigned* __restrict__ mbS,
                           int nm) {
  int p = blockIdx.x * 256 + threadIdx.x;
  unsigned bc = 0, bs = 0;
  for (int i = 0; i < nm; ++i) {
    if (cm[(long)i * HWP + p] > 0) bc |= 1u << i;
    if (sm[(long)i * HWP + p] > 0) bs |= 1u << i;
  }
  mbC[p] = bc;
  mbS[p] = bs;
}

// per-mask pixel counts (float), one block per mask
__global__ void k_counts(const unsigned* __restrict__ mb, float* __restrict__ cnt) {
  __shared__ float red[256];
  int i = blockIdx.x;
  float s = 0.f;
  for (int p = threadIdx.x; p < HWP; p += 256) s += (float)((mb[p] >> i) & 1u);
  red[threadIdx.x] = s;
  __syncthreads();
  for (int k = 128; k > 0; k >>= 1) {
    if (threadIdx.x < k) red[threadIdx.x] += red[threadIdx.x + k];
    __syncthreads();
  }
  if (threadIdx.x == 0) cnt[i] = red[0];
}

// per-(mask, channel) sums; grid = (C, nm); input is channel-major NCHW
__global__ void k_chansum(const float* __restrict__ x, const unsigned* __restrict__ mb,
                          float* __restrict__ sums) {
  __shared__ float red[256];
  int c = blockIdx.x, i = blockIdx.y;
  const float* row = x + (long)c * HWP;
  float s = 0.f;
  for (int p = threadIdx.x; p < HWP; p += 256)
    if ((mb[p] >> i) & 1u) s += row[p];
  red[threadIdx.x] = s;
  __syncthreads();
  for (int k = 128; k > 0; k >>= 1) {
    if (threadIdx.x < k) red[threadIdx.x] += red[threadIdx.x + k];
    __syncthreads();
  }
  if (threadIdx.x == 0) sums[(long)i * CC + c] = red[0];
}

// mean-subtracted features, f32 NCHW -> bf16 channel-last [HW][C].
// LDS tile transpose so both global reads and writes are coalesced.
__global__ __launch_bounds__(256) void k_feat(
    const float* __restrict__ x, const unsigned* __restrict__ mb,
    const float* __restrict__ sums, const float* __restrict__ cnt,
    bf16_t* __restrict__ feat, int nm) {
  __shared__ float t[32][33];
  const int p0 = blockIdx.x * 32;
  const int c0 = blockIdx.y * 32;
  const int tid = threadIdx.x;
  for (int e = tid; e < 1024; e += 256) {
    int pi = e & 31, ci = e >> 5;
    t[ci][pi] = x[(long)(c0 + ci) * HWP + p0 + pi];
  }
  __syncthreads();
  for (int e = tid; e < 1024; e += 256) {
    int ci = e & 31, pi = e >> 5;
    int p = p0 + pi;
    unsigned bits = mb[p];
    int j = -1;
    for (int i = 0; i < nm; ++i)
      if (((bits >> i) & 1u) && cnt[i] >= 10.f) j = i;
    float v = t[ci][pi];
    if (j >= 0) v -= sums[(long)j * CC + c0 + ci] / fmaxf(cnt[j], 1.f);
    feat[(long)p * CC + c0 + ci] = (bf16_t)v;
  }
}

// ---------------------------------------------------------------------------
// 3x3 conv (pad 1) as implicit GEMM with bf16 WMMA, channel-last activations.
// Block: 64 out-channels x 32 pixels (one row), 128 threads (4 waves).
// K-loop: 32-channel input chunks staged in LDS as [3 rows][34 x][32 ic]
// via async-to-LDS b128 copies (contiguous both sides).
// CMOUT: store output channel-major [OC][HW] (for the covariance kernel).
// ---------------------------------------------------------------------------
template <int IC, int OC, bool RELU, bool CMOUT>
__global__ __launch_bounds__(128) void k_conv(
    const bf16_t* __restrict__ in, const bf16_t* __restrict__ wt,
    const float* __restrict__ bias, bf16_t* __restrict__ out) {
  __shared__ __align__(16) bf16_t lds[3 * 34 * 32];
  const int tid  = threadIdx.x;
  const int lane = tid & 31;
  const int wave = tid >> 5;
  const int n    = lane & 15;
  const int hi   = lane >> 4;
  const int tileX = blockIdx.x & 7;   // W/32 = 8 tiles per row
  const int y     = blockIdx.x >> 3;
  const int x0    = tileX * 32;
  const int nt    = wave & 1;
  const int mbase = blockIdx.y * 64 + (wave >> 1) * 32;
  constexpr bool FULL = (OC % 64) == 0;
  const bool active = FULL || (mbase + 32 <= OC);  // mbase multiple of 32

  v8f c0 = {}; v8f c1 = {};

  for (int ic0 = 0; ic0 < IC; ic0 += 32) {
    __syncthreads();
    // stage input tile: 102 (row,x) positions x 4 chunks of 8 bf16 (16B)
    for (int e = tid; e < 102 * 4; e += 128) {
      int chunk = e & 3;
      int pos   = e >> 2;
      int xi    = pos % 34;
      int row   = pos / 34;
      int gx = x0 - 1 + xi;
      int gy = y - 1 + row;
      bf16_t* dst = &lds[(row * 34 + xi) * 32 + chunk * 8];
      if ((unsigned)gx < (unsigned)WW && (unsigned)gy < (unsigned)HH) {
        const bf16_t* src =
            in + ((long)gy * WW + gx) * IC + ic0 + chunk * 8;
        copy16_g2l(src, dst);
      } else {
#pragma unroll
        for (int z = 0; z < 8; ++z) dst[z] = (bf16_t)0.f;
      }
    }
    if (ic0 + 32 < IC)  // keep the next chunk warm in L2
      __builtin_prefetch(in + ((long)y * WW + x0) * IC + ic0 + 32, 0, 2);
    async_wait_all();
    __syncthreads();

    if (active) {
#pragma unroll
      for (int ky = 0; ky < 3; ++ky) {
        // preload 3 B fragments and 6 A fragments, then 6 WMMAs
        union { v16bf v; v8bf h[2]; } bfrag[3], a0[3], a1[3];
#pragma unroll
        for (int kx = 0; kx < 3; ++kx) {
          const bf16_t* bp = &lds[(ky * 34 + nt * 16 + n + kx) * 32];
          bfrag[kx].h[0] = *(const v8bf*)(bp + hi * 8);
          bfrag[kx].h[1] = *(const v8bf*)(bp + 16 + hi * 8);
          const bf16_t* w0 =
              wt + ((long)(mbase + n) * 9 + ky * 3 + kx) * IC + ic0;
          a0[kx].h[0] = *(const v8bf*)(w0 + hi * 8);
          a0[kx].h[1] = *(const v8bf*)(w0 + 16 + hi * 8);
          const bf16_t* w1 =
              wt + ((long)(mbase + 16 + n) * 9 + ky * 3 + kx) * IC + ic0;
          a1[kx].h[0] = *(const v8bf*)(w1 + hi * 8);
          a1[kx].h[1] = *(const v8bf*)(w1 + 16 + hi * 8);
        }
#pragma unroll
        for (int kx = 0; kx < 3; ++kx) {
          c0 = wmma_bf16(a0[kx].v, bfrag[kx].v, c0);
          c1 = wmma_bf16(a1[kx].v, bfrag[kx].v, c1);
        }
      }
    }
  }

  if (!active) return;
  const int px = x0 + nt * 16 + n;
  if (CMOUT) {
#pragma unroll
    for (int r = 0; r < 8; ++r) {
      int oc0 = mbase + r + hi * 8;
      int oc1 = mbase + 16 + r + hi * 8;
      float v0 = c0[r] + bias[oc0];
      float v1 = c1[r] + bias[oc1];
      if (RELU) { v0 = fmaxf(v0, 0.f); v1 = fmaxf(v1, 0.f); }
      out[(long)oc0 * HWP + (long)y * WW + px] = (bf16_t)v0;
      out[(long)oc1 * HWP + (long)y * WW + px] = (bf16_t)v1;
    }
  } else {
    union { v8bf v; bf16_t e[8]; } o0, o1;
#pragma unroll
    for (int r = 0; r < 8; ++r) {
      float v0 = c0[r] + bias[mbase + r + hi * 8];
      float v1 = c1[r] + bias[mbase + 16 + r + hi * 8];
      if (RELU) { v0 = fmaxf(v0, 0.f); v1 = fmaxf(v1, 0.f); }
      o0.e[r] = (bf16_t)v0;
      o1.e[r] = (bf16_t)v1;
    }
    bf16_t* ob = out + ((long)y * WW + px) * OC + mbase;
    *(v8bf*)(ob + hi * 8) = o0.v;
    *(v8bf*)(ob + 16 + hi * 8) = o1.v;
  }
}

// ---------------------------------------------------------------------------
// Masked covariance: cov[a][b] = sum_p m[p]*fv[a][p]*fv[b][p] (mask binary).
// Only the B operand needs masking (m^2 == m). fv is channel-major [32][HW];
// WMMA with K = pixels; atomicAdd partial tiles into cov[i].
// ---------------------------------------------------------------------------
__global__ __launch_bounds__(128) void k_cov(
    const bf16_t* __restrict__ fv, const unsigned* __restrict__ mbits,
    float* __restrict__ cov) {
  __shared__ unsigned char keep[128];
  const int tid  = threadIdx.x;
  const int lane = tid & 31;
  const int wave = tid >> 5;
  const int n    = lane & 15;
  const int hi   = lane >> 4;
  const int mt   = wave & 1;
  const int nt   = wave >> 1;
  const int i    = blockIdx.y;
  const int span = HWP / gridDim.x;
  const int pstart = blockIdx.x * span;

  v8f c = {};
  for (int pb = pstart; pb < pstart + span; pb += 128) {
    __syncthreads();
    keep[tid] = (unsigned char)((mbits[pb + tid] >> i) & 1u);
    __syncthreads();
#pragma unroll
    for (int s = 0; s < 4; ++s) {
      const int p0 = pb + s * 32;
      const unsigned char* kp = keep + s * 32;
      union { v16bf v; v8bf h[2]; } a;
      union { v16bf v; bf16_t e[16]; v8bf h[2]; } b;
      {
        const bf16_t* rp = fv + (long)(mt * 16 + n) * HWP + p0;
        a.h[0] = *(const v8bf*)(rp + hi * 16);
        a.h[1] = *(const v8bf*)(rp + hi * 16 + 8);
      }
      {
        const bf16_t* rp = fv + (long)(nt * 16 + n) * HWP + p0 + hi * 16;
        b.h[0] = *(const v8bf*)(rp);
        b.h[1] = *(const v8bf*)(rp + 8);
#pragma unroll
        for (int e = 0; e < 16; ++e)
          if (!kp[hi * 16 + e]) b.e[e] = (bf16_t)0.f;
      }
      c = wmma_bf16(a.v, b.v, c);
    }
  }
  float* cv = cov + (long)i * (MDIM * MDIM);
#pragma unroll
  for (int r = 0; r < 8; ++r) {
    int A = mt * 16 + r + hi * 8;
    int B = nt * 16 + n;
    atomicAdd(&cv[A * MDIM + B], c[r]);
  }
}

// fc on flattened covariance (divided by count); grid = (8, nm), 128 thr
__global__ void k_fc(const float* __restrict__ fcw, const float* __restrict__ fcb,
                     const float* __restrict__ cov, const float* __restrict__ cnt,
                     float* __restrict__ mats) {
  int i = blockIdx.y;
  int o = blockIdx.x * 128 + threadIdx.x;
  float inv = 1.f / fmaxf(cnt[i], 1.f);
  const float* cv = cov + (long)i * 1024;
  float acc = fcb[o];
  for (int k = 0; k < 1024; ++k) acc += fcw[(long)o * 1024 + k] * (cv[k] * inv);
  mats[(long)i * 1024 + o] = acc;
}

// tm_i = sMat_i @ cMat_i  (32x32x32); grid = (nm), 1024 thr
__global__ void k_tm(const float* __restrict__ matS, const float* __restrict__ matC,
                     float* __restrict__ tm) {
  int i = blockIdx.x;
  int a = threadIdx.x >> 5;
  int b = threadIdx.x & 31;
  const float* S = matS + (long)i * 1024;
  const float* C = matC + (long)i * 1024;
  float acc = 0.f;
  for (int k = 0; k < 32; ++k) acc += S[a * 32 + k] * C[k * 32 + b];
  tm[(long)i * 1024 + a * 32 + b] = acc;
}

// ---------------------------------------------------------------------------
// ccv = compress_w(32x512) @ featC + compress_b -> bf16 channel-last [HW][32]
// featC channel-last: async-staged LDS tile [32 px][32 ic], contiguous copies.
// ---------------------------------------------------------------------------
__global__ __launch_bounds__(128) void k_ccv(
    const bf16_t* __restrict__ feat, const bf16_t* __restrict__ cw,
    const float* __restrict__ cb, bf16_t* __restrict__ ccv) {
  __shared__ __align__(16) bf16_t ldsT[32 * 32];  // [px][ic]
  const int tid  = threadIdx.x;
  const int lane = tid & 31;
  const int wave = tid >> 5;
  const int n    = lane & 15;
  const int hi   = lane >> 4;
  const int mt   = wave & 1;
  const int nt   = wave >> 1;
  const int p0   = blockIdx.x * 32;

  v8f c = {};
  for (int k0 = 0; k0 < CC; k0 += 32) {
    __syncthreads();
    {
      int xp = tid >> 2, chunk = tid & 3;  // 32 px x 4 chunks = 128 copies
      copy16_g2l(feat + (long)(p0 + xp) * CC + k0 + chunk * 8,
                 &ldsT[xp * 32 + chunk * 8]);
    }
    async_wait_all();
    __syncthreads();
    union { v16bf v; v8bf h[2]; } a, b;
    {
      const bf16_t* bp = &ldsT[(nt * 16 + n) * 32 + hi * 16];
      b.h[0] = *(const v8bf*)(bp);
      b.h[1] = *(const v8bf*)(bp + 8);
    }
    {
      const bf16_t* wr = cw + (long)(mt * 16 + n) * CC + k0;
      a.h[0] = *(const v8bf*)(wr + hi * 16);
      a.h[1] = *(const v8bf*)(wr + hi * 16 + 8);
    }
    c = wmma_bf16(a.v, b.v, c);
  }
  const int px = p0 + nt * 16 + n;
  union { v8bf v; bf16_t e[8]; } o;
#pragma unroll
  for (int r = 0; r < 8; ++r) {
    int oc = mt * 16 + r + hi * 8;
    o.e[r] = (bf16_t)(c[r] + cb[oc]);
  }
  *(v8bf*)(ccv + (long)px * MDIM + mt * 16 + hi * 8) = o.v;
}

// ---------------------------------------------------------------------------
// Fused final: per 16-pixel tile, per-pixel t = tm_j @ ccv (or ccv),
// out = unzip_w @ t + unzip_b + sMean_j.   M=512, K=32, one WMMA per tile.
// ccv is channel-last [HW][32].
// ---------------------------------------------------------------------------
__global__ __launch_bounds__(128) void k_final(
    const bf16_t* __restrict__ ccv, const float* __restrict__ tmats,
    const bf16_t* __restrict__ uw, const float* __restrict__ ub,
    const unsigned* __restrict__ mbC, const float* __restrict__ cntC,
    const float* __restrict__ cntS, const float* __restrict__ sumsS,
    float* __restrict__ out, int nm) {
  __shared__ int jsel[16];
  __shared__ __align__(16) bf16_t tt[16 * 32];  // [px][k]
  const int tid  = threadIdx.x;
  const int lane = tid & 31;
  const int wave = tid >> 5;
  const int n    = lane & 15;
  const int hi   = lane >> 4;
  const int p0   = blockIdx.x * 16;

  if (tid < 16) {
    unsigned bits = mbC[p0 + tid];
    int j = -1;
    for (int i = 0; i < nm; ++i)
      if (((bits >> i) & 1u) && cntC[i] >= 10.f && cntS[i] >= 10.f) j = i;
    jsel[tid] = j;
  }
  __syncthreads();

  for (int e = tid; e < 512; e += 128) {
    int px = e >> 5;
    int k  = e & 31;
    int j  = jsel[px];
    const bf16_t* cv = ccv + (long)(p0 + px) * MDIM;
    float v;
    if (j >= 0) {
      const float* tm = tmats + (long)j * 1024 + k * 32;
      v = 0.f;
      for (int m2 = 0; m2 < 32; ++m2) v += tm[m2] * (float)cv[m2];
    } else {
      v = (float)cv[k];
    }
    tt[px * 32 + k] = (bf16_t)v;
  }
  __syncthreads();

  union { v16bf v; v8bf h[2]; } b;
  {
    const bf16_t* bp = &tt[n * 32 + hi * 16];
    b.h[0] = *(const v8bf*)(bp);
    b.h[1] = *(const v8bf*)(bp + 8);
  }
  const int px = p0 + n;
  const int j  = jsel[n];
  const float sInv = (j >= 0) ? 1.f / fmaxf(cntS[j], 1.f) : 0.f;

  for (int q = 0; q < 8; ++q) {
    int mtile = wave * 8 + q;  // 32 tiles -> 512 out channels
    union { v16bf v; v8bf h[2]; } a;
    const bf16_t* wr = uw + (long)(mtile * 16 + n) * 32;
    a.h[0] = *(const v8bf*)(wr + hi * 16);
    a.h[1] = *(const v8bf*)(wr + hi * 16 + 8);
    v8f c = {};
    c = wmma_bf16(a.v, b.v, c);
#pragma unroll
    for (int r = 0; r < 8; ++r) {
      int oc = mtile * 16 + r + hi * 8;
      float v = c[r] + ub[oc];
      if (j >= 0) v += sumsS[(long)j * CC + oc] * sInv;
      out[(long)oc * HWP + px] = v;
    }
  }
}

// ---------------------------------------------------------------------------
// Host orchestration
// ---------------------------------------------------------------------------
extern "C" void kernel_launch(void* const* d_in, const int* in_sizes, int n_in,
                              void* d_out, int out_size, void* d_ws, size_t ws_size,
                              hipStream_t stream) {
  const float* cF = (const float*)d_in[0];
  const float* sF = (const float*)d_in[1];
  const int* cm = (const int*)d_in[2];
  const int* sm = (const int*)d_in[3];
  const float* s_w1 = (const float*)d_in[4];  const float* s_b1 = (const float*)d_in[5];
  const float* s_w2 = (const float*)d_in[6];  const float* s_b2 = (const float*)d_in[7];
  const float* s_w3 = (const float*)d_in[8];  const float* s_b3 = (const float*)d_in[9];
  const float* s_fcw = (const float*)d_in[10]; const float* s_fcb = (const float*)d_in[11];
  const float* c_w1 = (const float*)d_in[12]; const float* c_b1 = (const float*)d_in[13];
  const float* c_w2 = (const float*)d_in[14]; const float* c_b2 = (const float*)d_in[15];
  const float* c_w3 = (const float*)d_in[16]; const float* c_b3 = (const float*)d_in[17];
  const float* c_fcw = (const float*)d_in[18]; const float* c_fcb = (const float*)d_in[19];
  const float* cmp_w = (const float*)d_in[20]; const float* cmp_b = (const float*)d_in[21];
  const float* uz_w  = (const float*)d_in[22]; const float* uz_b  = (const float*)d_in[23];
  float* out = (float*)d_out;

  int NM = in_sizes[2] / HWP;
  if (NM > NMAXM) NM = NMAXM;
  if (NM < 1) NM = 1;

  char* base = (char*)d_ws;
  size_t off = 0;
  auto alloc = [&](size_t b) {
    void* p = base + off;
    off += (b + 255) & ~(size_t)255;
    return p;
  };

  bf16_t* featC = (bf16_t*)alloc((size_t)CC * HWP * 2);   // [HW][512]
  bf16_t* featS = (bf16_t*)alloc((size_t)CC * HWP * 2);
  bf16_t* act1  = (bf16_t*)alloc((size_t)256 * HWP * 2);  // [HW][256]
  bf16_t* act2  = (bf16_t*)alloc((size_t)128 * HWP * 2);  // [HW][128]
  bf16_t* fvS   = (bf16_t*)alloc((size_t)32 * HWP * 2);   // [32][HW]
  bf16_t* fvC   = (bf16_t*)alloc((size_t)32 * HWP * 2);   // [32][HW]
  bf16_t* ccv   = (bf16_t*)alloc((size_t)32 * HWP * 2);   // [HW][32]
  bf16_t* wt1s  = (bf16_t*)alloc((size_t)256 * 512 * 9 * 2);
  bf16_t* wt1c  = (bf16_t*)alloc((size_t)256 * 512 * 9 * 2);
  bf16_t* wt2s  = (bf16_t*)alloc((size_t)128 * 256 * 9 * 2);
  bf16_t* wt2c  = (bf16_t*)alloc((size_t)128 * 256 * 9 * 2);
  bf16_t* wt3s  = (bf16_t*)alloc((size_t)32 * 128 * 9 * 2);
  bf16_t* wt3c  = (bf16_t*)alloc((size_t)32 * 128 * 9 * 2);
  bf16_t* cwb   = (bf16_t*)alloc((size_t)32 * 512 * 2);
  bf16_t* uwb   = (bf16_t*)alloc((size_t)512 * 32 * 2);
  unsigned* mbC = (unsigned*)alloc((size_t)HWP * 4);
  unsigned* mbS = (unsigned*)alloc((size_t)HWP * 4);
  float* sumsC  = (float*)alloc((size_t)NMAXM * CC * 4);
  float* sumsS  = (float*)alloc((size_t)NMAXM * CC * 4);
  float* cntC   = (float*)alloc((size_t)NMAXM * 4);
  float* cntS   = (float*)alloc((size_t)NMAXM * 4);
  float* covC   = (float*)alloc((size_t)NMAXM * 1024 * 4);
  float* covS   = (float*)alloc((size_t)NMAXM * 1024 * 4);
  float* matC   = (float*)alloc((size_t)NMAXM * 1024 * 4);
  float* matS   = (float*)alloc((size_t)NMAXM * 1024 * 4);
  float* tmats  = (float*)alloc((size_t)NMAXM * 1024 * 4);
  (void)ws_size; (void)n_in; (void)out_size;

  // weights -> bf16
  auto pack = [&](const float* w, bf16_t* wt, int OC, int IC) {
    long total = (long)OC * IC * 9;
    k_packconv<<<dim3((unsigned)((total + 255) / 256)), dim3(256), 0, stream>>>(w, wt, OC, IC);
  };
  pack(s_w1, wt1s, 256, 512); pack(s_w2, wt2s, 128, 256); pack(s_w3, wt3s, 32, 128);
  pack(c_w1, wt1c, 256, 512); pack(c_w2, wt2c, 128, 256); pack(c_w3, wt3c, 32, 128);
  k_cast<<<dim3(64), dim3(256), 0, stream>>>(cmp_w, cwb, 32 * 512);
  k_cast<<<dim3(64), dim3(256), 0, stream>>>(uz_w, uwb, 512 * 32);

  // masks / stats
  k_maskbits<<<dim3(HWP / 256), dim3(256), 0, stream>>>(cm, sm, mbC, mbS, NM);
  k_counts<<<dim3(NM), dim3(256), 0, stream>>>(mbC, cntC);
  k_counts<<<dim3(NM), dim3(256), 0, stream>>>(mbS, cntS);
  k_chansum<<<dim3(CC, NM), dim3(256), 0, stream>>>(cF, mbC, sumsC);
  k_chansum<<<dim3(CC, NM), dim3(256), 0, stream>>>(sF, mbS, sumsS);

  // mean-subtracted features (bf16, channel-last)
  k_feat<<<dim3(HWP / 32, CC / 32), dim3(256), 0, stream>>>(cF, mbC, sumsC, cntC, featC, NM);
  k_feat<<<dim3(HWP / 32, CC / 32), dim3(256), 0, stream>>>(sF, mbS, sumsS, cntS, featS, NM);

  // zero covariance accumulators
  k_zerof<<<dim3((NM * 1024 + 255) / 256), dim3(256), 0, stream>>>(covC, NM * 1024);
  k_zerof<<<dim3((NM * 1024 + 255) / 256), dim3(256), 0, stream>>>(covS, NM * 1024);

  const dim3 cb128(128);
  // s-net
  k_conv<512, 256, true, false><<<dim3(HWP / 32, 4), cb128, 0, stream>>>(featS, wt1s, s_b1, act1);
  k_conv<256, 128, true, false><<<dim3(HWP / 32, 2), cb128, 0, stream>>>(act1, wt2s, s_b2, act2);
  k_conv<128, 32, false, true><<<dim3(HWP / 32, 1), cb128, 0, stream>>>(act2, wt3s, s_b3, fvS);
  k_cov<<<dim3(64, NM), cb128, 0, stream>>>(fvS, mbS, covS);

  // c-net
  k_conv<512, 256, true, false><<<dim3(HWP / 32, 4), cb128, 0, stream>>>(featC, wt1c, c_b1, act1);
  k_conv<256, 128, true, false><<<dim3(HWP / 32, 2), cb128, 0, stream>>>(act1, wt2c, c_b2, act2);
  k_conv<128, 32, false, true><<<dim3(HWP / 32, 1), cb128, 0, stream>>>(act2, wt3c, c_b3, fvC);
  k_cov<<<dim3(64, NM), cb128, 0, stream>>>(fvC, mbC, covC);

  // fc + tm
  k_fc<<<dim3(8, NM), cb128, 0, stream>>>(s_fcw, s_fcb, covS, cntS, matS);
  k_fc<<<dim3(8, NM), cb128, 0, stream>>>(c_fcw, c_fcb, covC, cntC, matC);
  k_tm<<<dim3(NM), dim3(1024), 0, stream>>>(matS, matC, tmats);

  // compress + fused transform/unzip
  k_ccv<<<dim3(HWP / 32), cb128, 0, stream>>>(featC, cwb, cmp_b, ccv);
  k_final<<<dim3(HWP / 16), cb128, 0, stream>>>(ccv, tmats, uwb, uz_b, mbC,
                                                cntC, cntS, sumsS, out, NM);
}
